// InstanceGCN_30013231464488
// MI455X (gfx1250) — compile-verified, matrix operand order checked by reading, not run on hardware
//
#include <hip/hip_runtime.h>
#include <hip/hip_bf16.h>

// ---------------------------------------------------------------------------
// InstanceGCN on MI455X (gfx1250): bandwidth-bound hetero-GNN.
// Dense linear layers use v_wmma_f32_16x16x32_f16 (H=10 padded into 16x16x32).
// All dense-tile global I/O is coalesced float4 via LDS staging; fragment
// assembly uses clamped-index selects (no exec-mask serialization).
// Edge message passing = per-edge float2 gather + global_atomic_add_f32
// scatter with both rsqrt-degree normalizations folded into one scale.
// ---------------------------------------------------------------------------

typedef __attribute__((ext_vector_type(16))) _Float16 v16h;
typedef __attribute__((ext_vector_type(8)))  float    v8f;

#define HD 10   // hidden width
#define FZD 16  // z input features

__device__ __forceinline__ v8f wmma_f16(v16h a, v16h b, v8f c) {
  // (neg_a, A, neg_b, B, c_mod, C, reuse_a, reuse_b)
  return __builtin_amdgcn_wmma_f32_16x16x32_f16(false, a, false, b, (short)0, c,
                                                false, false);
}

// 16x32 f16 fragment from a 16x16 row-major LDS tile t[i][k]; k>=16 -> 0.
// A-matrix: lane holds row i = lane&15; K base 0 (lanes 0-15) / 8 (lanes 16-31);
// regs 0..7 = K kb..kb+7, regs 8..15 = K kb+16.. (zero, K<16 here).
// Same loader serves B when the tile is stored transposed (t[n][k] = B[k][n]).
__device__ __forceinline__ v16h frag_from_lds16(const _Float16* t, int lane) {
  int i  = lane & 15;
  int kb = (lane & 16) ? 8 : 0;
  v16h a;
#pragma unroll
  for (int j = 0; j < 8; ++j) a[j] = t[i * 16 + kb + j];
#pragma unroll
  for (int j = 8; j < 16; ++j) a[j] = (_Float16)0.f;
  return a;
}

// A fragment from an LDS f32 tile of 16 rows x HD cols (row-major, stride HD).
// Branch-free: clamped index + value select (ds_load + v_cndmask).
__device__ __forceinline__ v16h fragA_lds_h10(const float* rows16, int lane) {
  int i  = lane & 15;
  int kb = (lane & 16) ? 8 : 0;
  const float* rp = rows16 + i * HD;
  v16h a;
#pragma unroll
  for (int j = 0; j < 8; ++j) {
    int k  = kb + j;
    int kk = (k < HD) ? k : 0;
    float v = rp[kk];
    a[j] = (_Float16)((k < HD) ? v : 0.f);
  }
#pragma unroll
  for (int j = 8; j < 16; ++j) a[j] = (_Float16)0.f;
  return a;
}

// ---------------------------------------------------------------------------

__global__ void k_zero4(float4* __restrict__ p, long n4) {
  long i      = (long)blockIdx.x * blockDim.x + threadIdx.x;
  long stride = (long)gridDim.x * blockDim.x;
  float4 z = {0.f, 0.f, 0.f, 0.f};
  for (; i < n4; i += stride) p[i] = z;
}

__global__ void k_count(const int* __restrict__ idx, int* __restrict__ cnt, int n) {
  int i = blockIdx.x * blockDim.x + threadIdx.x;
  if (i < n) atomicAdd(&cnt[idx[i]], 1);
}

__global__ void k_rsqrt_deg(const int* __restrict__ cnt, float* __restrict__ rs, int n) {
  int i = blockIdx.x * blockDim.x + threadIdx.x;
  if (i < n) {
    int c = cnt[i];
    rs[i] = rsqrtf((float)(c > 0 ? c : 1));
  }
}

// h_z = relu(z_feat @ Wz + bz): 128 rows/block, one WMMA per 16-row wave tile.
__global__ __launch_bounds__(256) void k_encode_z(const float* __restrict__ z,
                                                  const float* __restrict__ Wz,
                                                  const float* __restrict__ bz,
                                                  float* __restrict__ hz, int n) {
  __shared__ _Float16 Bt[256];            // Bt[n][k] = Wz[k][n], zero-padded
  __shared__ float tin[128 * FZD];        // 8 KB input stage
  __shared__ float tout[128 * HD];        // 5 KB output stage
  int t = threadIdx.x;
  {
    int nn = t >> 4, k = t & 15;
    Bt[t] = (_Float16)((nn < HD) ? Wz[k * HD + nn] : 0.f);
  }
  int row0 = blockIdx.x * 128;
  bool full = (row0 + 128 <= n);          // block-uniform -> scalar branch
  if (full) {
    const float4* src = (const float4*)(z + (long)row0 * FZD);
    float4* dst = (float4*)tin;
    for (int idx = t; idx < 128 * FZD / 4; idx += 256) dst[idx] = src[idx];
  } else {
    for (int idx = t; idx < 128 * FZD; idx += 256) {
      int rr = row0 + (idx >> 4);
      tin[idx] = (rr < n) ? z[(long)row0 * FZD + idx] : 0.f;
    }
  }
  __syncthreads();

  int lane = t & 31, wave = t >> 5;
  int i = lane & 15, kb = (lane & 16) ? 8 : 0;
  const float* rp = &tin[(wave * 16 + i) * FZD];
  v16h a;
#pragma unroll
  for (int j = 0; j < 8; ++j) a[j] = (_Float16)rp[kb + j];   // K=16, all valid
#pragma unroll
  for (int j = 8; j < 16; ++j) a[j] = (_Float16)0.f;

  v16h b = frag_from_lds16(Bt, lane);
  int ncol = lane & 15, hi = (lane >> 4) & 1;
  float bias = (ncol < HD) ? bz[ncol] : 0.f;
  v8f c;
#pragma unroll
  for (int r = 0; r < 8; ++r) c[r] = bias;
  v8f d = wmma_f16(a, b, c);

  if (ncol < HD) {
#pragma unroll
    for (int r = 0; r < 8; ++r)
      tout[(wave * 16 + r + 8 * hi) * HD + ncol] = fmaxf(d[r], 0.f);
  }
  __syncthreads();
  if (full) {
    const float4* src = (const float4*)tout;
    float4* dst = (float4*)(hz + (long)row0 * HD);
    for (int idx = t; idx < 128 * HD / 4; idx += 256) dst[idx] = src[idx];
  } else {
    for (int idx = t; idx < 128 * HD; idx += 256) {
      int rr = row0 + idx / HD;
      if (rr < n) hz[(long)row0 * HD + idx] = tout[idx];
    }
  }
}

// h_x = relu(x * Wx + bx)  (rank-1 input); float2 row stores.
__global__ void k_encode_x(const float* __restrict__ x, const float* __restrict__ Wx,
                           const float* __restrict__ bx, float* __restrict__ hx, int n) {
  int i = blockIdx.x * blockDim.x + threadIdx.x;
  if (i < n) {
    float v = x[i];
    float2* hp = (float2*)hx + (long)i * (HD / 2);
#pragma unroll
    for (int j = 0; j < HD / 2; ++j) {
      float2 o;
      o.x = fmaxf(v * Wx[2 * j]     + bx[2 * j],     0.f);
      o.y = fmaxf(v * Wx[2 * j + 1] + bx[2 * j + 1], 0.f);
      hp[j] = o;
    }
  }
}

// agg[dst] += h[src] * rs_src[src] * rs_dst[dst]   (both norms folded per edge)
__global__ void k_scatter(const int* __restrict__ src, const int* __restrict__ dst,
                          const float* __restrict__ hsrc,
                          const float* __restrict__ rs_src,
                          const float* __restrict__ rs_dst,
                          float* __restrict__ agg, int ne) {
  int e = blockIdx.x * blockDim.x + threadIdx.x;
  if (e < ne) {
    int s = src[e], d = dst[e];
    float sc = rs_src[s] * rs_dst[d];
    const float2* hp = (const float2*)hsrc + (long)s * (HD / 2);  // 8B-aligned rows
    float* ap = agg + (long)d * HD;
#pragma unroll
    for (int j = 0; j < HD / 2; ++j) {
      float2 v = hp[j];
      atomicAdd(&ap[2 * j],     v.x * sc);
      atomicAdd(&ap[2 * j + 1], v.y * sc);
    }
  }
}

// out = relu(agg @ W + bias_scale*b); LDS-staged coalesced I/O, WMMA compute.
__global__ __launch_bounds__(256) void k_conv(const float* __restrict__ agg,
                                              const float* __restrict__ W,
                                              const float* __restrict__ bvec,
                                              float bias_scale,
                                              float* __restrict__ out, int n) {
  __shared__ _Float16 Bt[256];
  __shared__ float tin[128 * HD];
  __shared__ float tout[128 * HD];
  int t = threadIdx.x;
  {
    int nn = t >> 4, k = t & 15;
    Bt[t] = (_Float16)((nn < HD && k < HD) ? W[k * HD + nn] : 0.f);
  }
  int row0 = blockIdx.x * 128;
  bool full = (row0 + 128 <= n);
  if (full) {
    const float4* src = (const float4*)(agg + (long)row0 * HD);
    float4* dst = (float4*)tin;
    for (int idx = t; idx < 128 * HD / 4; idx += 256) dst[idx] = src[idx];
  } else {
    for (int idx = t; idx < 128 * HD; idx += 256) {
      int rr = row0 + idx / HD;
      tin[idx] = (rr < n) ? agg[(long)row0 * HD + idx] : 0.f;
    }
  }
  __syncthreads();

  int lane = t & 31, wave = t >> 5;
  v16h a = fragA_lds_h10(&tin[wave * 16 * HD], lane);
  v16h b = frag_from_lds16(Bt, lane);
  int ncol = lane & 15, hi = (lane >> 4) & 1;
  float bias = (ncol < HD) ? bvec[ncol] * bias_scale : 0.f;
  v8f c;
#pragma unroll
  for (int r = 0; r < 8; ++r) c[r] = bias;
  v8f d = wmma_f16(a, b, c);

  if (ncol < HD) {
#pragma unroll
    for (int r = 0; r < 8; ++r)
      tout[(wave * 16 + r + 8 * hi) * HD + ncol] = fmaxf(d[r], 0.f);
  }
  __syncthreads();
  if (full) {
    const float4* src = (const float4*)tout;
    float4* dst = (float4*)(out + (long)row0 * HD);
    for (int idx = t; idx < 128 * HD / 4; idx += 256) dst[idx] = src[idx];
  } else {
    for (int idx = t; idx < 128 * HD; idx += 256) {
      int rr = row0 + idx / HD;
      if (rr < n) out[(long)row0 * HD + idx] = tout[idx];
    }
  }
}

// Fused 3-layer MLP + segment-sum readout. Three chained WMMAs per wave tile;
// D->A relayout through per-wave f16 LDS stage tiles.
__global__ __launch_bounds__(256) void k_mlp_readout(
    const float* __restrict__ hz2,
    const float* __restrict__ Wo1, const float* __restrict__ bo1,
    const float* __restrict__ Wo2, const float* __restrict__ bo2,
    const float* __restrict__ Wo3, const float* __restrict__ bo3,
    const int* __restrict__ gid, float* __restrict__ sums, int n) {
  __shared__ _Float16 B1[256], B2[256], B3[256];
  __shared__ float tin[128 * HD];
  __shared__ _Float16 stage[8][256];
  int t = threadIdx.x;
  {
    int nn = t >> 4, k = t & 15;
    B1[t] = (_Float16)((nn < HD && k < HD) ? Wo1[k * HD + nn] : 0.f);
    B2[t] = (_Float16)((nn < HD && k < HD) ? Wo2[k * HD + nn] : 0.f);
    B3[t] = (_Float16)((nn == 0 && k < HD) ? Wo3[k] : 0.f);
  }
  int row0 = blockIdx.x * 128;
  bool full = (row0 + 128 <= n);
  if (full) {
    const float4* src = (const float4*)(hz2 + (long)row0 * HD);
    float4* dst = (float4*)tin;
    for (int idx = t; idx < 128 * HD / 4; idx += 256) dst[idx] = src[idx];
  } else {
    for (int idx = t; idx < 128 * HD; idx += 256) {
      int rr = row0 + idx / HD;
      tin[idx] = (rr < n) ? hz2[(long)row0 * HD + idx] : 0.f;
    }
  }
  __syncthreads();

  int lane = t & 31, wave = t >> 5;
  int ncol = lane & 15, hi = (lane >> 4) & 1;
  v8f c, d;

  // layer 1: relu(hz2 @ Wo1 + bo1)
  v16h a = fragA_lds_h10(&tin[wave * 16 * HD], lane);
  v16h b = frag_from_lds16(B1, lane);
  float bias = (ncol < HD) ? bo1[ncol] : 0.f;
#pragma unroll
  for (int r = 0; r < 8; ++r) c[r] = bias;
  d = wmma_f16(a, b, c);
#pragma unroll
  for (int r = 0; r < 8; ++r)
    stage[wave][(r + 8 * hi) * 16 + ncol] = (_Float16)fmaxf(d[r], 0.f);
  __syncthreads();

  // layer 2: relu(t1 @ Wo2 + bo2)
  a = frag_from_lds16(&stage[wave][0], lane);
  b = frag_from_lds16(B2, lane);
  bias = (ncol < HD) ? bo2[ncol] : 0.f;
#pragma unroll
  for (int r = 0; r < 8; ++r) c[r] = bias;
  d = wmma_f16(a, b, c);
  __syncthreads();
#pragma unroll
  for (int r = 0; r < 8; ++r)
    stage[wave][(r + 8 * hi) * 16 + ncol] = (_Float16)fmaxf(d[r], 0.f);
  __syncthreads();

  // layer 3: t2 @ Wo3 + bo3 (no relu); column 0 holds the logit
  a = frag_from_lds16(&stage[wave][0], lane);
  b = frag_from_lds16(B3, lane);
  bias = (ncol == 0) ? bo3[0] : 0.f;
#pragma unroll
  for (int r = 0; r < 8; ++r) c[r] = bias;
  d = wmma_f16(a, b, c);

  if (ncol == 0) {  // lanes 0 and 16 hold rows r and r+8
#pragma unroll
    for (int r = 0; r < 8; ++r) {
      int rr = row0 + wave * 16 + r + 8 * hi;
      if (rr < n) atomicAdd(&sums[gid[rr]], d[r]);
    }
  }
}

__global__ void k_finalize(const float* __restrict__ sums,
                           const int* __restrict__ cntB,
                           float* __restrict__ out, int nb) {
  int i = blockIdx.x * blockDim.x + threadIdx.x;
  if (i < nb) out[i] = sums[i] / fmaxf((float)cntB[i], 1.f);
}

// ---------------------------------------------------------------------------

extern "C" void kernel_launch(void* const* d_in, const int* in_sizes, int n_in,
                              void* d_out, int out_size, void* d_ws, size_t ws_size,
                              hipStream_t stream) {
  const float* z_feat  = (const float*)d_in[0];
  const float* x_feat  = (const float*)d_in[1];
  // d_in[2] = c_feat: dead (initial h_c is overwritten before use)
  const int* src_z2c = (const int*)d_in[3];
  const int* dst_z2c = (const int*)d_in[4];
  const int* src_x2c = (const int*)d_in[5];
  const int* dst_x2c = (const int*)d_in[6];
  const int* zgid    = (const int*)d_in[7];
  const float* Wz  = (const float*)d_in[8];
  const float* bz  = (const float*)d_in[9];
  const float* Wx  = (const float*)d_in[10];
  const float* bx  = (const float*)d_in[11];
  // d_in[12..15] = Wc,bc,Wg1,bg1: dead (Wg1 results overwritten by Wg2)
  const float* Wg2 = (const float*)d_in[16];
  const float* bg2 = (const float*)d_in[17];
  const float* Wo1 = (const float*)d_in[18];
  const float* bo1 = (const float*)d_in[19];
  const float* Wo2 = (const float*)d_in[20];
  const float* bo2 = (const float*)d_in[21];
  const float* Wo3 = (const float*)d_in[22];
  const float* bo3 = (const float*)d_in[23];

  const int NZ = in_sizes[7];       // 400k
  const int NX = in_sizes[1];       // 400k
  const int NC = in_sizes[2];       // 800k
  const int E  = in_sizes[3];       // 8M
  const int NB = out_size;          // 1024

  // Workspace layout (floats); zero-init region first. ~147 MB total.
  float* ws = (float*)d_ws;
  size_t off = 0;
  float* aggC  = ws + off; off += (size_t)NC * HD;
  float* aggZ  = ws + off; off += (size_t)NZ * HD;
  int*   cntZ  = (int*)(ws + off); off += NZ;
  int*   cntCz = (int*)(ws + off); off += NC;
  int*   cntX  = (int*)(ws + off); off += NX;
  int*   cntCx = (int*)(ws + off); off += NC;
  float* sums  = ws + off; off += NB;
  int*   cntB  = (int*)(ws + off); off += NB;
  const size_t zeroElems = off;
  float* rsz  = ws + off; off += NZ;
  float* rscz = ws + off; off += NC;
  float* rsx  = ws + off; off += NX;
  float* rscx = ws + off; off += NC;
  float* hz   = ws + off; off += (size_t)NZ * HD;
  float* hx   = ws + off; off += (size_t)NX * HD;
  float* hc   = ws + off; off += (size_t)NC * HD;
  float* hz2  = ws + off; off += (size_t)NZ * HD;

  dim3 blk(256);
  // 1. zero accumulators / counters (spillover <4 floats lands in rsz, rewritten)
  k_zero4<<<2048, blk, 0, stream>>>((float4*)ws, (long)((zeroElems + 3) / 4));

  // 2. degree counts + graph-size histogram
  int gE = (E + 255) / 256;
  k_count<<<gE, blk, 0, stream>>>(src_z2c, cntZ, E);
  k_count<<<gE, blk, 0, stream>>>(dst_z2c, cntCz, E);
  k_count<<<gE, blk, 0, stream>>>(src_x2c, cntX, E);
  k_count<<<gE, blk, 0, stream>>>(dst_x2c, cntCx, E);
  k_count<<<(NZ + 255) / 256, blk, 0, stream>>>(zgid, cntB, NZ);
  k_rsqrt_deg<<<(NZ + 255) / 256, blk, 0, stream>>>(cntZ, rsz, NZ);
  k_rsqrt_deg<<<(NC + 255) / 256, blk, 0, stream>>>(cntCz, rscz, NC);
  k_rsqrt_deg<<<(NX + 255) / 256, blk, 0, stream>>>(cntX, rsx, NX);
  k_rsqrt_deg<<<(NC + 255) / 256, blk, 0, stream>>>(cntCx, rscx, NC);

  // 3. node encoders
  k_encode_z<<<(NZ + 127) / 128, blk, 0, stream>>>(z_feat, Wz, bz, hz, NZ);
  k_encode_x<<<(NX + 255) / 256, blk, 0, stream>>>(x_feat, Wx, bx, hx, NX);

  // 4. forward message passing into c (both convs share Wg2 -> bias counted 2x)
  k_scatter<<<gE, blk, 0, stream>>>(src_z2c, dst_z2c, hz, rsz, rscz, aggC, E);
  k_scatter<<<gE, blk, 0, stream>>>(src_x2c, dst_x2c, hx, rsx, rscx, aggC, E);
  k_conv<<<(NC + 127) / 128, blk, 0, stream>>>(aggC, Wg2, bg2, 2.0f, hc, NC);

  // 5. reverse message passing c -> z
  k_scatter<<<gE, blk, 0, stream>>>(dst_z2c, src_z2c, hc, rscz, rsz, aggZ, E);
  k_conv<<<(NZ + 127) / 128, blk, 0, stream>>>(aggZ, Wg2, bg2, 1.0f, hz2, NZ);

  // 6. fused MLP + per-graph sum, then mean
  k_mlp_readout<<<(NZ + 127) / 128, blk, 0, stream>>>(hz2, Wo1, bo1, Wo2, bo2,
                                                      Wo3, bo3, zgid, sums, NZ);
  k_finalize<<<(NB + 255) / 256, blk, 0, stream>>>(sums, cntB, (float*)d_out, NB);
}
